// TestModel_53042846105764
// MI455X (gfx1250) — 8-bit matrix operand order repaired and checked by reading, compiled, not run on hardware
//
#include <hip/hip_runtime.h>


#define HDIM 4096
#define NROW 8192
#define FP8_MAX 448.0f

typedef __attribute__((ext_vector_type(16))) int      v16i;
typedef __attribute__((ext_vector_type(8)))  float    v8f;
typedef __attribute__((ext_vector_type(4)))  unsigned u32x4;
typedef __attribute__((ext_vector_type(8)))  int      i32x8;
typedef __attribute__((ext_vector_type(4)))  int      i32x4;

#if defined(__gfx1250__) && __has_builtin(__builtin_amdgcn_tensor_load_to_lds) && \
    __has_builtin(__builtin_amdgcn_s_wait_tensorcnt)
#define USE_TDM 1
#else
#define USE_TDM 0
#endif

__device__ inline unsigned f32_to_e4m3_sw(float f) {
  float c = fminf(fmaxf(f, -FP8_MAX), FP8_MAX);
  unsigned u = __float_as_uint(c);
  unsigned sign = (u >> 31) << 7;
  unsigned a = u & 0x7FFFFFFFu;
  unsigned r;
  if (a < 0x3C800000u) {
    float af = __uint_as_float(a);
    r = (unsigned)(int)rintf(af * 512.0f);
  } else {
    unsigned lsb = (a >> 20) & 1u;
    a += 0x7FFFFu + lsb;
    unsigned e = a >> 23;
    unsigned m = (a >> 20) & 7u;
    r = ((e - 120u) << 3) | m;
  }
  return sign | r;
}

__device__ inline unsigned pack4_e4m3(float a, float b, float c, float d) {
  a = fminf(fmaxf(a, -FP8_MAX), FP8_MAX);
  b = fminf(fmaxf(b, -FP8_MAX), FP8_MAX);
  c = fminf(fmaxf(c, -FP8_MAX), FP8_MAX);
  d = fminf(fmaxf(d, -FP8_MAX), FP8_MAX);
#if defined(__gfx1250__) && __has_builtin(__builtin_amdgcn_cvt_pk_fp8_f32)
  int p = __builtin_amdgcn_cvt_pk_fp8_f32(a, b, 0, false);
  p     = __builtin_amdgcn_cvt_pk_fp8_f32(c, d, p, true);
  return (unsigned)p;
#else
  return f32_to_e4m3_sw(a) | (f32_to_e4m3_sw(b) << 8) |
         (f32_to_e4m3_sw(c) << 16) | (f32_to_e4m3_sw(d) << 24);
#endif
}

__device__ inline void tdm_load_2d(const unsigned char* g, unsigned char* l,
                                   unsigned tile_d0, unsigned tile_d1,
                                   unsigned tensor_d0, unsigned tensor_d1,
                                   unsigned stride) {
#if USE_TDM
  unsigned long long ga = (unsigned long long)(size_t)g;
  unsigned lo = (unsigned)(size_t)(void*)l;
  u32x4 g0;
  g0[0] = 1u;
  g0[1] = lo;
  g0[2] = (unsigned)ga;
  g0[3] = (unsigned)(ga >> 32) | 0x80000000u;
  i32x8 g1;
  g1[0] = 0;
  g1[1] = (int)((tensor_d0 & 0xFFFFu) << 16);
  g1[2] = (int)((tensor_d0 >> 16) | ((tensor_d1 & 0xFFFFu) << 16));
  g1[3] = (int)((tensor_d1 >> 16) | (tile_d0 << 16));
  g1[4] = (int)(tile_d1 & 0xFFFFu);
  g1[5] = (int)stride;
  g1[6] = 0;
  g1[7] = 0;
  i32x4 z4 = {0, 0, 0, 0};
#if __clang_major__ >= 23
  i32x8 z8 = {0, 0, 0, 0, 0, 0, 0, 0};
  __builtin_amdgcn_tensor_load_to_lds(g0, g1, z4, z4, z8, 0);
#else
  __builtin_amdgcn_tensor_load_to_lds(g0, g1, z4, z4, 0);
#endif
#else
  (void)g; (void)l; (void)tile_d0; (void)tile_d1;
  (void)tensor_d0; (void)tensor_d1; (void)stride;
#endif
}

__device__ inline float blk_reduce(float val, float* red, bool domax) {
  const int tid = threadIdx.x;
  red[tid] = val;
  __syncthreads();
#pragma unroll
  for (int s = 128; s > 0; s >>= 1) {
    if (tid < s)
      red[tid] = domax ? fmaxf(red[tid], red[tid + s]) : (red[tid] + red[tid + s]);
    __syncthreads();
  }
  float r = red[0];
  __syncthreads();
  return r;
}

template <int MODE>
__global__ __launch_bounds__(256) void rowop_kernel(
    const float* __restrict__ in, float* __restrict__ out,
    const float* __restrict__ nw, unsigned char* __restrict__ qx,
    float* __restrict__ sx) {
  __shared__ float red[256];
  const int tid = threadIdx.x;
  const long base = (long)blockIdx.x * HDIM;

  const float4* inrow = (const float4*)(in + base);
  float v[16];
  float ss = 0.0f;
#pragma unroll
  for (int i = 0; i < 4; ++i) {
    float4 f = inrow[tid + 256 * i];
    if (MODE == 0) {
      f.x = fmaxf(f.x, 0.0f); f.y = fmaxf(f.y, 0.0f);
      f.z = fmaxf(f.z, 0.0f); f.w = fmaxf(f.w, 0.0f);
      ((float4*)(out + base))[tid + 256 * i] = f;
    }
    v[4 * i + 0] = f.x; v[4 * i + 1] = f.y;
    v[4 * i + 2] = f.z; v[4 * i + 3] = f.w;
    ss += f.x * f.x + f.y * f.y + f.z * f.z + f.w * f.w;
  }

  const float tot = blk_reduce(ss, red, false);
  const float inv = rsqrtf(tot * (1.0f / HDIM) + 1e-6f);

  float amax = 0.0f;
#pragma unroll
  for (int i = 0; i < 4; ++i) {
    float4 w4 = ((const float4*)nw)[tid + 256 * i];
    v[4 * i + 0] *= inv * w4.x; v[4 * i + 1] *= inv * w4.y;
    v[4 * i + 2] *= inv * w4.z; v[4 * i + 3] *= inv * w4.w;
    amax = fmaxf(amax, fmaxf(fmaxf(fabsf(v[4 * i + 0]), fabsf(v[4 * i + 1])),
                             fmaxf(fabsf(v[4 * i + 2]), fabsf(v[4 * i + 3]))));
  }

  if (MODE == 2) {
#pragma unroll
    for (int i = 0; i < 4; ++i) {
      float4 o = {v[4 * i + 0], v[4 * i + 1], v[4 * i + 2], v[4 * i + 3]};
      ((float4*)(out + base))[tid + 256 * i] = o;
    }
    return;
  }

  const float am = blk_reduce(amax, red, true);
  const float scale = fmaxf(am * (1.0f / FP8_MAX), 1e-12f);
  if (tid == 0) sx[blockIdx.x] = scale;
  const float qs = 1.0f / scale;

  unsigned* qrow = (unsigned*)(qx + base);
#pragma unroll
  for (int i = 0; i < 4; ++i)
    qrow[tid + 256 * i] = pack4_e4m3(v[4 * i + 0] * qs, v[4 * i + 1] * qs,
                                     v[4 * i + 2] * qs, v[4 * i + 3] * qs);
}

__global__ __launch_bounds__(256) void wamax_part_kernel(
    const float* __restrict__ W, float* __restrict__ part) {
  __shared__ float red[256];
  const int n4 = (HDIM * HDIM) / 4;
  const float4* w4 = (const float4*)W;
  float m = 0.0f;
  for (int i = blockIdx.x * blockDim.x + threadIdx.x; i < n4;
       i += gridDim.x * blockDim.x) {
    float4 f = w4[i];
    m = fmaxf(m, fmaxf(fmaxf(fabsf(f.x), fabsf(f.y)),
                       fmaxf(fabsf(f.z), fabsf(f.w))));
  }
  m = blk_reduce(m, red, true);
  if (threadIdx.x == 0) part[blockIdx.x] = m;
}

__global__ __launch_bounds__(256) void wamax_final_kernel(
    const float* __restrict__ part, int n, float* __restrict__ swp) {
  __shared__ float red[256];
  float m = 0.0f;
  for (int i = threadIdx.x; i < n; i += 256) m = fmaxf(m, part[i]);
  m = blk_reduce(m, red, true);
  if (threadIdx.x == 0) swp[0] = fmaxf(m * (1.0f / FP8_MAX), 1e-12f);
}

__global__ __launch_bounds__(256) void wquant_kernel(
    const float* __restrict__ W, const float* __restrict__ swp,
    unsigned char* __restrict__ qw) {
  const float qs = 1.0f / swp[0];
  const int n4 = (HDIM * HDIM) / 4;
  const float4* w4 = (const float4*)W;
  unsigned* q4 = (unsigned*)qw;
  for (int i = blockIdx.x * blockDim.x + threadIdx.x; i < n4;
       i += gridDim.x * blockDim.x) {
    float4 f = w4[i];
    q4[i] = pack4_e4m3(f.x * qs, f.y * qs, f.z * qs, f.w * qs);
  }
}

extern __shared__ unsigned char smem[];

__global__ __launch_bounds__(256) void fp8_gemm_resid_kernel(
    const unsigned char* __restrict__ qx, const unsigned char* __restrict__ qw,
    const float* __restrict__ sx, const float* __restrict__ swp,
    float* __restrict__ resid) {
  const int tid  = threadIdx.x;
  const int wave = tid >> 5;
  const int lane = tid & 31;
  const int half = lane >> 4;
  const int l16  = lane & 15;
  const long rowBase = (long)blockIdx.y * 128;
  const long colBase = (long)blockIdx.x * 128;

  v8f acc[8];
#pragma unroll
  for (int j = 0; j < 8; ++j)
#pragma unroll
    for (int e = 0; e < 8; ++e) acc[j][e] = 0.0f;

  const int cr = tid >> 3;
  const int ck = (tid & 7) << 4;

  auto stage = [&](int k0, int b) {
    unsigned char* dA = smem + (b << 15);
    unsigned char* dB = dA + 16384;
#if USE_TDM
    tdm_load_2d(qx + (rowBase + 16 * wave) * (long)HDIM + k0,
                dA + 16 * wave * 128, 128, 16, HDIM, NROW, HDIM);
    tdm_load_2d(qw + (colBase + 16 * wave) * (long)HDIM + k0,
                dB + 16 * wave * 128, 128, 16, HDIM, HDIM, HDIM);
#else
#pragma unroll
    for (int i = 0; i < 4; ++i) {
      const int r = cr + 32 * i;
      *(uint4*)(dA + r * 128 + ck) =
          *(const uint4*)(qx + (rowBase + r) * (long)HDIM + k0 + ck);
      *(uint4*)(dB + r * 128 + ck) =
          *(const uint4*)(qw + (colBase + r) * (long)HDIM + k0 + ck);
    }
#endif
  };

  stage(0, 0);

  for (int it = 0; it < HDIM / 128; ++it) {
    const int cur = it & 1;
    if (it + 1 < HDIM / 128) stage((it + 1) * 128, cur ^ 1);
#if USE_TDM
    if (it + 1 < HDIM / 128) __builtin_amdgcn_s_wait_tensorcnt(2);
    else                     __builtin_amdgcn_s_wait_tensorcnt(0);
#endif
    __syncthreads();

    const unsigned char* curA = smem + (cur << 15);
    const unsigned char* curB = curA + 16384;

    v16i afrag;
    {
      const unsigned char* ab = curA + (16 * wave + l16) * 128 + half * 16;
#pragma unroll
      for (int i = 0; i < 8; ++i) {
        int2 d = *(const int2*)(ab + 32 * (i >> 1) + 8 * (i & 1));
        afrag[2 * i]     = d.x;
        afrag[2 * i + 1] = d.y;
      }
    }

#pragma unroll
    for (int j = 0; j < 8; ++j) {
      v16i bfrag;
      const unsigned char* bb = curB + (16 * j + l16) * 128 + half * 16;
#pragma unroll
      for (int i = 0; i < 4; ++i) {
        int4 d = *(const int4*)(bb + 32 * i);
        bfrag[4 * i + 0] = d.x; bfrag[4 * i + 1] = d.y;
        bfrag[4 * i + 2] = d.z; bfrag[4 * i + 3] = d.w;
      }
      acc[j] = __builtin_amdgcn_wmma_f32_16x16x128_fp8_fp8(
          afrag, bfrag, (short)0, acc[j], false, false);
    }
    __syncthreads();
  }

  const float sw = swp[0];
  float srow[8];
#pragma unroll
  for (int v = 0; v < 8; ++v)
    srow[v] = sx[rowBase + 16 * wave + half * 8 + v] * sw;

#pragma unroll
  for (int j = 0; j < 8; ++j) {
    const long col = colBase + 16 * j + l16;
#pragma unroll
    for (int v = 0; v < 8; ++v) {
      const long row = rowBase + 16 * wave + half * 8 + v;
      float* p = resid + row * (long)HDIM + col;
      *p = *p + acc[j][v] * srow[v];
    }
  }
}

extern "C" void kernel_launch(void* const* d_in, const int* in_sizes, int n_in,
                              void* d_out, int out_size, void* d_ws,
                              size_t ws_size, hipStream_t stream) {
  const float* x = (const float*)d_in[0];
  const float* nw[4] = {(const float*)d_in[1], (const float*)d_in[2],
                        (const float*)d_in[3], (const float*)d_in[4]};
  const float* W[3] = {(const float*)d_in[5], (const float*)d_in[6],
                       (const float*)d_in[7]};
  float* resid = (float*)d_out;

  unsigned char* ws = (unsigned char*)d_ws;
  unsigned char* qx = ws;
  unsigned char* qw = ws + (size_t)NROW * HDIM;
  float* sx   = (float*)(ws + (size_t)NROW * HDIM + (size_t)HDIM * HDIM);
  float* part = sx + NROW;
  float* swp  = part + 1024;

  const dim3 gemmGrid(HDIM / 128, NROW / 128);

  rowop_kernel<0><<<NROW, 256, 0, stream>>>(x, resid, nw[0], qx, sx);

  for (int l = 0; l < 3; ++l) {
    wamax_part_kernel<<<1024, 256, 0, stream>>>(W[l], part);
    wamax_final_kernel<<<1, 256, 0, stream>>>(part, 1024, swp);
    wquant_kernel<<<4096, 256, 0, stream>>>(W[l], swp, qw);
    fp8_gemm_resid_kernel<<<gemmGrid, 256, 65536, stream>>>(qx, qw, sx, swp,
                                                            resid);
    if (l < 2)
      rowop_kernel<1><<<NROW, 256, 0, stream>>>(resid, nullptr, nw[l + 1], qx, sx);
    else
      rowop_kernel<2><<<NROW, 256, 0, stream>>>(resid, resid, nw[3], nullptr,
                                                nullptr);
  }
}
